// ConvBlock_10101763080279
// MI455X (gfx1250) — matrix operand order repaired and checked by reading, not compiled, not run on hardware
//
#include <hip/hip_runtime.h>
#include <hip/hip_bf16.h>

typedef __attribute__((ext_vector_type(16))) _Float16 v16h;
typedef __attribute__((ext_vector_type(8)))  _Float16 v8h;
typedef __attribute__((ext_vector_type(8)))  float    v8f;

// Problem constants
#define NBR   6          // branches (IN_C * MODES)
#define NFEAT 64
#define FEAT_KS 360      // LDS channel stride per pixel (mult of 8 for 16B-aligned frags)
#define PATCH_OFF 328    // conv input patch lives at channels [328,360)
#define TILE_PIX 64      // pixels per workgroup
#define NTILES 512       // 32768 pixels per branch / 64

// Workspace layout (units: _Float16 halves)
#define XIMG_OFF 0
#define XIMG_SZ  (2*8*2*66*66)          // 139392
#define W1P_OFF  (XIMG_OFF + XIMG_SZ)
#define W1P_SZ   (NBR*64*32)            // conv, K padded 18->32
#define W2P_OFF  (W1P_OFF + W1P_SZ)
#define W2P_SZ   (NBR*64*64)
#define W3P_OFF  (W2P_OFF + W2P_SZ)
#define W3P_SZ   (NBR*64*128)
#define W4P_OFF  (W3P_OFF + W3P_SZ)
#define W4P_SZ   (NBR*64*192)
#define W5P_OFF  (W4P_OFF + W4P_SZ)
#define W5P_SZ   (NBR*64*256)
#define W6P_OFF  (W5P_OFF + W5P_SZ)
#define W6P_SZ   (NBR*16*320)           // M padded 8->16
#define PACK_TOTAL (W1P_SZ+W2P_SZ+W3P_SZ+W4P_SZ+W5P_SZ+W6P_SZ)

// ---------------------------------------------------------------------------
// Prep 1: rotated + edge-padded inputs to f16. ximg[c][rot*2+b][cur/prev][66][66]
// ---------------------------------------------------------------------------
__global__ void build_ximg(const float* __restrict__ x, const float* __restrict__ px,
                           _Float16* __restrict__ ximg) {
  int idx = blockIdx.x * blockDim.x + threadIdx.x;
  if (idx >= XIMG_SZ) return;
  int wp  = idx % 66;  int t1 = idx / 66;
  int hp  = t1 % 66;   int t2 = t1 / 66;
  int ci  = t2 & 1;    int t3 = t2 >> 1;
  int img = t3 & 7;    int c  = t3 >> 3;
  int r = img >> 1, b = img & 1;
  int h = hp < 63 ? hp : 63;     // edge pad (replicate) bottom/right
  int w = wp < 63 ? wp : 63;
  int sh, sw;                    // rot90(x,k=r): xr[i,j] = x[...]
  switch (r) {
    case 0:  sh = h;      sw = w;      break;
    case 1:  sh = w;      sw = 63 - h; break;
    case 2:  sh = 63 - h; sw = 63 - w; break;
    default: sh = 63 - w; sw = h;      break;
  }
  const float* src = (ci == 0) ? x : px;
  ximg[idx] = (_Float16)src[((b * 2 + c) * 64 + sh) * 64 + sw];
}

// ---------------------------------------------------------------------------
// Prep 2: pack all weights to f16 with padding.
// ---------------------------------------------------------------------------
__global__ void pack_weights(const float* __restrict__ W1, const float* __restrict__ W2,
                             const float* __restrict__ W3, const float* __restrict__ W4,
                             const float* __restrict__ W5, const float* __restrict__ W6,
                             _Float16* __restrict__ ws) {
  int idx = blockIdx.x * blockDim.x + threadIdx.x;
  if (idx >= PACK_TOTAL) return;
  int i = idx;
  if (i < W1P_SZ) {                      // [NBR][64][32], src [NBR][64][2][3][3]
    int k = i & 31; int o = (i >> 5) & 63; int nb = i >> 11;
    float v = 0.f;
    if (k < 18) v = W1[((nb * 64 + o) * 2 + (k / 9)) * 9 + (k % 9)];
    ws[W1P_OFF + i] = (_Float16)v; return;
  }
  i -= W1P_SZ;
  if (i < W2P_SZ) { ws[W2P_OFF + i] = (_Float16)W2[i]; return; }
  i -= W2P_SZ;
  if (i < W3P_SZ) { ws[W3P_OFF + i] = (_Float16)W3[i]; return; }
  i -= W3P_SZ;
  if (i < W4P_SZ) { ws[W4P_OFF + i] = (_Float16)W4[i]; return; }
  i -= W4P_SZ;
  if (i < W5P_SZ) { ws[W5P_OFF + i] = (_Float16)W5[i]; return; }
  i -= W5P_SZ;
  {                                      // [NBR][16][320], src [NBR][8][320]
    int k = i % 320; int m = (i / 320) & 15; int nb = i / (320 * 16);
    float v = (m < 8) ? W6[(nb * 8 + m) * 320 + k] : 0.f;
    ws[W6P_OFF + i] = (_Float16)v;
  }
}

// ---------------------------------------------------------------------------
// Fragment loaders (ISA 7.12.2 layouts, wave32)
// A 16x32 f16: lanes 0-15 hold M=lane, K {0..7,16..23}; lanes 16-31 K {8..15,24..31}
// B 32x16 f16: lanes 0-15 hold N=lane, K {0..7,16..23}; lanes 16-31 K {8..15,24..31}
// ---------------------------------------------------------------------------
__device__ __forceinline__ v16h load_a_frag(const _Float16* __restrict__ W, int KSw,
                                            int row0, int kc, int lm, int hi) {
  const _Float16* p = W + (row0 + lm) * KSw + kc * 32 + hi * 8;
  union { v16h v; v8h h[2]; } u;
  u.h[0] = *(const v8h*)p;
  u.h[1] = *(const v8h*)(p + 16);
  return u.v;
}
__device__ __forceinline__ v16h load_b_frag(const _Float16* feat, int nt, int kc,
                                            int kbase, int lm, int hi) {
  const _Float16* p = feat + (nt * 16 + lm) * FEAT_KS + kbase + kc * 32 + hi * 8;
  union { v16h v; v8h h[2]; } u;
  u.h[0] = *(const v8h*)p;
  u.h[1] = *(const v8h*)(p + 16);
  return u.v;
}

// One dense layer: out[m,n] = relu(W[m,:K] @ feat[:K,n] + bias[m]), appended at cbOut.
// Wave -> (mt = wave/2, two N-tiles). A frag shared between both N-tiles.
__device__ __forceinline__ void dense_layer(_Float16* feat, const _Float16* __restrict__ W,
                                            const float* __restrict__ bias,
                                            int K, int kbase, int cbOut,
                                            int wave, int lm, int hi) {
  const int mt  = wave >> 1;
  const int nt0 = (wave & 1) * 2, nt1 = nt0 + 1;
  v8f acc0 = {}; v8f acc1 = {};
  const int nk = K >> 5;
  for (int kc = 0; kc < nk; ++kc) {
    v16h a  = load_a_frag(W, K, mt * 16, kc, lm, hi);
    v16h b0 = load_b_frag(feat, nt0, kc, kbase, lm, hi);
    acc0 = __builtin_amdgcn_wmma_f32_16x16x32_f16(false, a, false, b0, (short)0, acc0, false, false);
    v16h b1 = load_b_frag(feat, nt1, kc, kbase, lm, hi);
    acc1 = __builtin_amdgcn_wmma_f32_16x16x32_f16(false, a, false, b1, (short)0, acc1, false, false);
  }
  union { v8f v; float f[8]; } c0, c1; c0.v = acc0; c1.v = acc1;
#pragma unroll
  for (int v = 0; v < 8; ++v) {           // C layout: VGPR v -> M = v + hi*8
    int m = mt * 16 + hi * 8 + v;
    float bv = bias[m];
    float f0 = c0.f[v] + bv; f0 = f0 > 0.f ? f0 : 0.f;
    float f1 = c1.f[v] + bv; f1 = f1 > 0.f ? f1 : 0.f;
    feat[(nt0 * 16 + lm) * FEAT_KS + cbOut + m] = (_Float16)f0;
    feat[(nt1 * 16 + lm) * FEAT_KS + cbOut + m] = (_Float16)f1;
  }
  __syncthreads();                        // publish new channels for next layer
}

// ---------------------------------------------------------------------------
// Fused branch kernel: block = (branch nb, 64-pixel tile), 256 threads (8 waves)
// ---------------------------------------------------------------------------
__global__ __launch_bounds__(256) void mulut_fused(
    const _Float16* __restrict__ wsh,
    const float* __restrict__ B1, const float* __restrict__ B2,
    const float* __restrict__ B3, const float* __restrict__ B4,
    const float* __restrict__ B5, const float* __restrict__ B6,
    float* __restrict__ out) {
  __shared__ _Float16 feat[TILE_PIX * FEAT_KS];

  const int nb   = blockIdx.y;
  const int tile = blockIdx.x;
  const int tid  = threadIdx.x;
  const int wave = tid >> 5;
  const int lane = tid & 31;
  const int lm   = lane & 15;
  const int hi   = (lane >> 4) & 1;

  // Stage 0: build 3x3x2 conv patches (K padded to 32) for the 64 pixels.
  {
    int n  = tid >> 2;                       // pixel 0..63
    int kg = (tid & 3) << 3;                 // k group base
    int gid = tile * TILE_PIX + n;
    int img = gid >> 12, hh = (gid >> 6) & 63, ww = gid & 63;
    int c = nb / 3;                          // branch -> input channel (MODES=3)
    const _Float16* ximg = wsh + XIMG_OFF;
#pragma unroll
    for (int kk = 0; kk < 8; ++kk) {
      int k = kg + kk;
      _Float16 v = (_Float16)0.f;
      if (k < 18) {
        int ci = k / 9, r2 = k % 9, kh = r2 / 3, kw = r2 % 3;
        v = ximg[((c * 8 + img) * 2 + ci) * (66 * 66) + (hh + kh) * 66 + (ww + kw)];
      }
      feat[n * FEAT_KS + PATCH_OFF + k] = v;
    }
  }
  __syncthreads();

  // conv (as GEMM over padded K=32) + dense chain; features concatenated in LDS.
  dense_layer(feat, wsh + W1P_OFF + nb * 64 * 32,  B1 + nb * 64,  32, PATCH_OFF,   0, wave, lm, hi);
  dense_layer(feat, wsh + W2P_OFF + nb * 64 * 64,  B2 + nb * 64,  64, 0,          64, wave, lm, hi);
  dense_layer(feat, wsh + W3P_OFF + nb * 64 * 128, B3 + nb * 64, 128, 0,         128, wave, lm, hi);
  dense_layer(feat, wsh + W4P_OFF + nb * 64 * 192, B4 + nb * 64, 192, 0,         192, wave, lm, hi);
  dense_layer(feat, wsh + W5P_OFF + nb * 64 * 256, B5 + nb * 64, 256, 0,         256, wave, lm, hi);

  // Final 16x320 GEMM (M padded from 8) + pixel shuffle + un-rotate + quantize.
  if (wave < 4) {
    const int nt = wave;
    const _Float16* W6p = wsh + W6P_OFF + nb * 16 * 320;
    v8f acc = {};
    for (int kc = 0; kc < 10; ++kc) {
      v16h a = load_a_frag(W6p, 320, 0, kc, lm, hi);
      v16h b = load_b_frag(feat, nt, kc, 0, lm, hi);
      acc = __builtin_amdgcn_wmma_f32_16x16x32_f16(false, a, false, b, (short)0, acc, false, false);
    }
    if (!hi) {                               // lanes 0-15 hold M=0..7 == all 8 outputs
      union { v8f v; float f[8]; } cz; cz.v = acc;
      int n   = nt * 16 + lm;
      int gid = tile * TILE_PIX + n;
      int img = gid >> 12, hh = (gid >> 6) & 63, ww = gid & 63;
      int r = img >> 1, bb = img & 1;
#pragma unroll
      for (int ch = 0; ch < 8; ++ch) {
        float v = cz.f[ch] + B6[nb * 8 + ch];
        float contrib = rintf(tanhf(v) * 127.0f) * 0.5f;   // fold /IN_C
        int oc = ch >> 2, dr = (ch >> 1) & 1, dc = ch & 1;  // pixel shuffle
        int HH = 2 * hh + dr, WW = 2 * ww + dc;
        int i, j;                                           // inverse rot90^r, N=128
        switch (r) {
          case 0:  i = HH;       j = WW;       break;
          case 1:  i = WW;       j = 127 - HH; break;
          case 2:  i = 127 - HH; j = 127 - WW; break;
          default: i = 127 - WW; j = HH;       break;
        }
        atomicAdd(out + ((bb * 2 + oc) * 128 + i) * 128 + j, contrib);
      }
    }
  }
}

// ---------------------------------------------------------------------------
extern "C" void kernel_launch(void* const* d_in, const int* in_sizes, int n_in,
                              void* d_out, int out_size, void* d_ws, size_t ws_size,
                              hipStream_t stream) {
  const float* x  = (const float*)d_in[0];
  const float* px = (const float*)d_in[1];
  const float* W1 = (const float*)d_in[2];  const float* B1 = (const float*)d_in[3];
  const float* W2 = (const float*)d_in[4];  const float* B2 = (const float*)d_in[5];
  const float* W3 = (const float*)d_in[6];  const float* B3 = (const float*)d_in[7];
  const float* W4 = (const float*)d_in[8];  const float* B4 = (const float*)d_in[9];
  const float* W5 = (const float*)d_in[10]; const float* B5 = (const float*)d_in[11];
  const float* W6 = (const float*)d_in[12]; const float* B6 = (const float*)d_in[13];
  _Float16* wsh = (_Float16*)d_ws;
  float* out = (float*)d_out;

  hipMemsetAsync(out, 0, (size_t)out_size * sizeof(float), stream);  // capture-safe
  build_ximg<<<(XIMG_SZ + 255) / 256, 256, 0, stream>>>(x, px, wsh + XIMG_OFF);
  pack_weights<<<(PACK_TOTAL + 255) / 256, 256, 0, stream>>>(W1, W2, W3, W4, W5, W6, wsh);
  mulut_fused<<<dim3(NTILES, NBR), 256, 0, stream>>>(wsh, B1, B2, B3, B4, B5, B6, out);
}
